// MoELayer_33809982554646
// MI455X (gfx1250) — matrix operand order repaired and checked by reading, not compiled, not run on hardware
//
#include <hip/hip_runtime.h>
#include <stdint.h>

// ---- problem constants (from reference) ----
#define NE   8        // experts
#define TK   2        // top-k
#define DM   1024     // hidden
#define IM   4096     // intermediate
#define NT   8192     // tokens = 4*2048
#define PP   17408    // padded pair rows: NT*TK + NE*128
#define RB_  128      // row tile (pair rows)
#define CB_  64       // col tile
#define KB_  32       // K step (WMMA bf16 K)
#define LDK  (KB_+8)  // LDS row stride in elements -> 80 bytes (16B multiple)

typedef __attribute__((ext_vector_type(16))) __bf16       v16bf;
typedef __attribute__((ext_vector_type(8)))  float        v8f;
typedef __attribute__((ext_vector_type(4)))  float        fv4;
typedef __attribute__((ext_vector_type(4)))  unsigned int uv4;
typedef __attribute__((ext_vector_type(4)))  __bf16       bv4;
typedef __attribute__((ext_vector_type(2)))  __bf16       bv2;

struct Pair32 { uv4 lo, hi; };

// Build a 16-element bf16 WMMA fragment from two 16-byte LDS chunks.
static __device__ inline v16bf frag16(const __bf16* p0, const __bf16* p1) {
    Pair32 p;
    p.lo = *(const uv4*)p0;
    p.hi = *(const uv4*)p1;
    return __builtin_bit_cast(v16bf, p);
}

// ---------------- init: counters + padded-row token map ----------------
__global__ __launch_bounds__(256) void k_init(int* cnt, int* cursor, int* rowTok) {
    int i = blockIdx.x * 256 + threadIdx.x;
    if (i < NE) { cnt[i] = 0; cursor[i] = 0; }
    if (i < PP) rowTok[i] = -1;
}

// ---------------- x fp32 -> bf16 ----------------
__global__ __launch_bounds__(256) void k_cvt(const float* __restrict__ x, __bf16* __restrict__ xb) {
    size_t i = ((size_t)blockIdx.x * 256 + threadIdx.x) * 4;
    fv4 v = *(const fv4*)(x + i);
    bv4 b = { (__bf16)v.x, (__bf16)v.y, (__bf16)v.z, (__bf16)v.w };
    *(bv4*)(xb + i) = b;
}

// ---------------- router: logits, softmax, top-2, counts ----------------
__global__ __launch_bounds__(256) void k_router(const float* __restrict__ x,
                                                const float* __restrict__ wgate,
                                                float* __restrict__ probs,
                                                int2* __restrict__ tkIdx,
                                                float2* __restrict__ tkW,
                                                int* __restrict__ cnt) {
    __shared__ float wl[NE * DM];   // 32 KB
    int tid = threadIdx.x;
    for (int i = tid; i < NE * DM; i += 256) wl[i] = wgate[i];
    __syncthreads();

    int wave = tid >> 5, lane = tid & 31;
    int t = blockIdx.x * 8 + wave;
    const float* xr = x + (size_t)t * DM;

    float acc[NE];
    #pragma unroll
    for (int e = 0; e < NE; ++e) acc[e] = 0.f;
    for (int d = lane; d < DM; d += 32) {
        float xv = xr[d];
        #pragma unroll
        for (int e = 0; e < NE; ++e) acc[e] += xv * wl[e * DM + d];
    }
    #pragma unroll
    for (int e = 0; e < NE; ++e)
        #pragma unroll
        for (int m = 16; m >= 1; m >>= 1) acc[e] += __shfl_xor(acc[e], m, 32);

    if (lane == 0) {
        float mx = acc[0];
        #pragma unroll
        for (int e = 1; e < NE; ++e) mx = fmaxf(mx, acc[e]);
        float p[NE], s = 0.f;
        #pragma unroll
        for (int e = 0; e < NE; ++e) { p[e] = __expf(acc[e] - mx); s += p[e]; }
        float inv = 1.f / s;
        #pragma unroll
        for (int e = 0; e < NE; ++e) { p[e] *= inv; probs[(size_t)t * NE + e] = p[e]; }
        // top-2, ties -> lower index (matches jax top_k)
        int e0 = 0;
        #pragma unroll
        for (int e = 1; e < NE; ++e) if (p[e] > p[e0]) e0 = e;
        int e1 = (e0 == 0) ? 1 : 0;
        #pragma unroll
        for (int e = 0; e < NE; ++e) if (e != e0 && p[e] > p[e1]) e1 = e;
        float w0 = p[e0], w1 = p[e1], si = 1.f / (w0 + w1);
        tkIdx[t] = make_int2(e0, e1);
        tkW[t]   = make_float2(w0 * si, w1 * si);
        atomicAdd(&cnt[e0], 1);
        atomicAdd(&cnt[e1], 1);
    }
}

// ---------------- per-expert padded segment offsets ----------------
__global__ void k_scan(const int* cnt, int* padOff) {
    int off = 0;
    for (int e = 0; e < NE; ++e) { padOff[e] = off; off += ((cnt[e] + RB_ - 1) / RB_) * RB_; }
    padOff[NE] = off;
}

// ---------------- aux loss (deterministic fixed-order reduce) ----------------
__global__ __launch_bounds__(256) void k_aux(const float* __restrict__ probs,
                                             const int* __restrict__ cnt,
                                             float* __restrict__ auxOut) {
    __shared__ float red[256];
    int tid = threadIdx.x;
    float ps[NE];
    #pragma unroll
    for (int e = 0; e < NE; ++e) ps[e] = 0.f;
    for (int t = tid; t < NT; t += 256)
        #pragma unroll
        for (int e = 0; e < NE; ++e) ps[e] += probs[(size_t)t * NE + e];
    float aux = 0.f;
    for (int e = 0; e < NE; ++e) {
        red[tid] = ps[e];
        __syncthreads();
        for (int s = 128; s > 0; s >>= 1) {
            if (tid < s) red[tid] += red[tid + s];
            __syncthreads();
        }
        if (tid == 0) {
            float f = (float)cnt[e] / (float)(NT * TK);
            float p = red[0] / (float)NT;
            aux += f * p;
        }
        __syncthreads();
    }
    if (tid == 0) auxOut[0] = 0.02f * (float)NE * aux;
}

// ---------------- assign pair-rows grouped by expert ----------------
__global__ __launch_bounds__(256) void k_scatter(const int2* __restrict__ tkIdx,
                                                 const int* __restrict__ padOff,
                                                 int* __restrict__ cursor,
                                                 int* __restrict__ rowTok,
                                                 int* __restrict__ slotOf) {
    int t = blockIdx.x * 256 + threadIdx.x;
    if (t >= NT) return;
    int2 e = tkIdx[t];
    int s0 = atomicAdd(&cursor[e.x], 1); int p0 = padOff[e.x] + s0;
    rowTok[p0] = t; slotOf[2 * t] = p0;
    int s1 = atomicAdd(&cursor[e.y], 1); int p1 = padOff[e.y] + s1;
    rowTok[p1] = t; slotOf[2 * t + 1] = p1;
}

// Stage an 8-element (2 k-rows x 4 cols) fp32->bf16 transposed chunk into LDS
// using packed 32-bit stores (k, k+1 adjacent in [n][k] layout).
static __device__ inline void stageB(__bf16 (*bt)[LDK], const float* src,
                                     int pitch, int kp, int n0) {
    fv4 r0 = *(const fv4*)(src);
    fv4 r1 = *(const fv4*)(src + pitch);
    bv2 t0 = { (__bf16)r0.x, (__bf16)r1.x };
    bv2 t1 = { (__bf16)r0.y, (__bf16)r1.y };
    bv2 t2 = { (__bf16)r0.z, (__bf16)r1.z };
    bv2 t3 = { (__bf16)r0.w, (__bf16)r1.w };
    *(bv2*)&bt[n0 + 0][kp] = t0;
    *(bv2*)&bt[n0 + 1][kp] = t1;
    *(bv2*)&bt[n0 + 2][kp] = t2;
    *(bv2*)&bt[n0 + 3][kp] = t3;
}

// ---------------- GEMM1+2 fused: H = silu(X*Wg) * (X*Wu), bf16 out ----------------
__global__ __launch_bounds__(256) void k_gemm12(const __bf16* __restrict__ xb,
                                                const float*  __restrict__ wg,
                                                const float*  __restrict__ wu,
                                                const int*    __restrict__ rowTok,
                                                const int*    __restrict__ padOff,
                                                __bf16*       __restrict__ H) {
    const int rowStart = blockIdx.y * RB_;
    if (rowStart >= padOff[NE]) return;
    int e = 0;
    #pragma unroll
    for (int q = 1; q < NE; ++q) if (rowStart >= padOff[q]) e = q;
    const int colStart = blockIdx.x * CB_;
    const float* wgE = wg + (size_t)e * DM * IM;
    const float* wuE = wu + (size_t)e * DM * IM;

    __shared__ __bf16 aT[RB_][LDK];   // gathered token rows, [m][k]
    __shared__ __bf16 bgT[CB_][LDK];  // gate weights, transposed [n][k]
    __shared__ __bf16 buT[CB_][LDK];  // up weights,   transposed [n][k]

    const int tid  = threadIdx.x;
    const int lane = tid & 31;
    const int lm   = lane & 15;
    const int lsel = lane >> 4;
    const int wv   = tid >> 5;
    const int wr   = wv >> 1;   // wave row 0..3 (32 rows each)
    const int wc   = wv & 1;    // wave col 0..1 (32 cols each)

    const int arow  = tid >> 1;
    const int ahalf = tid & 1;
    const int atok  = rowTok[rowStart + arow];
    const int bkp   = (tid >> 4) * 2;    // even k: 0..30
    const int bn0   = (tid & 15) * 4;    // 0..60

    v8f accG[2][2], accU[2][2];
    #pragma unroll
    for (int i = 0; i < 2; ++i)
        #pragma unroll
        for (int j = 0; j < 2; ++j) {
            v8f z = {0.f,0.f,0.f,0.f,0.f,0.f,0.f,0.f};
            accG[i][j] = z; accU[i][j] = z;
        }

    for (int kk = 0; kk < DM; kk += KB_) {
        // stage A (zero rows for padding)
        uv4 a0 = {0,0,0,0}, a1 = {0,0,0,0};
        if (atok >= 0) {
            const uv4* src = (const uv4*)(xb + (size_t)atok * DM + kk + ahalf * 16);
            a0 = src[0]; a1 = src[1];
        }
        *(uv4*)&aT[arow][ahalf * 16]     = a0;
        *(uv4*)&aT[arow][ahalf * 16 + 8] = a1;
        // stage B: coalesced fp32 reads, transposed packed bf16-pair stores
        stageB(bgT, wgE + (size_t)(kk + bkp) * IM + colStart + bn0, IM, bkp, bn0);
        stageB(buT, wuE + (size_t)(kk + bkp) * IM + colStart + bn0, IM, bkp, bn0);
        __syncthreads();
        // fragments in contiguous k order (same for A and B): A 16x32 (lane m, 16 contiguous K at 16*lsel),
        // B 32x16 (lane n, 16 contiguous K at 16*lsel)
        v16bf af[2], bgf[2], buf2[2];
        #pragma unroll
        for (int i = 0; i < 2; ++i) {
            const __bf16* ap = &aT[wr * 32 + i * 16 + lm][lsel * 16];
            af[i] = frag16(ap, ap + 8);
        }
        #pragma unroll
        for (int j = 0; j < 2; ++j) {
            const __bf16* bp = &bgT[wc * 32 + j * 16 + lm][lsel * 16];
            bgf[j] = frag16(bp, bp + 8);
            const __bf16* bq = &buT[wc * 32 + j * 16 + lm][lsel * 16];
            buf2[j] = frag16(bq, bq + 8);
        }
        #pragma unroll
        for (int i = 0; i < 2; ++i)
            #pragma unroll
            for (int j = 0; j < 2; ++j) {
                accG[i][j] = __builtin_amdgcn_wmma_f32_16x16x32_bf16(
                    false, af[i], false, bgf[j], (short)0, accG[i][j], false, false);
                accU[i][j] = __builtin_amdgcn_wmma_f32_16x16x32_bf16(
                    false, af[i], false, buf2[j], (short)0, accU[i][j], false, false);
            }
        __syncthreads();
    }
    // epilogue: SwiGLU via fast rcp, store bf16 (C layout: m = r + 8*lsel, n = lm)
    #pragma unroll
    for (int i = 0; i < 2; ++i)
        #pragma unroll
        for (int j = 0; j < 2; ++j)
            #pragma unroll
            for (int r = 0; r < 8; ++r) {
                int m  = r + 8 * lsel;
                int gr = rowStart + wr * 32 + i * 16 + m;
                int gc = colStart + wc * 32 + j * 16 + lm;
                float g = accG[i][j][r];
                float u = accU[i][j][r];
                float h = g * __builtin_amdgcn_rcpf(1.f + __expf(-g)) * u; // silu(g)*u
                H[(size_t)gr * IM + gc] = (__bf16)h;
            }
}

// ---------------- GEMM3: out_pairs = H * Wd (fp32 partials) ----------------
__global__ __launch_bounds__(256) void k_gemm3(const __bf16* __restrict__ H,
                                               const float*  __restrict__ wd,
                                               const int*    __restrict__ padOff,
                                               float*        __restrict__ outp) {
    const int rowStart = blockIdx.y * RB_;
    if (rowStart >= padOff[NE]) return;
    int e = 0;
    #pragma unroll
    for (int q = 1; q < NE; ++q) if (rowStart >= padOff[q]) e = q;
    const int colStart = blockIdx.x * CB_;
    const float* wdE = wd + (size_t)e * IM * DM;

    __shared__ __bf16 aT[RB_][LDK];
    __shared__ __bf16 bT[CB_][LDK];

    const int tid  = threadIdx.x;
    const int lane = tid & 31;
    const int lm   = lane & 15;
    const int lsel = lane >> 4;
    const int wv   = tid >> 5;
    const int wr   = wv >> 1;
    const int wc   = wv & 1;

    const int arow  = tid >> 1;
    const int ahalf = tid & 1;
    const int bkp   = (tid >> 4) * 2;
    const int bn0   = (tid & 15) * 4;

    v8f acc[2][2];
    #pragma unroll
    for (int i = 0; i < 2; ++i)
        #pragma unroll
        for (int j = 0; j < 2; ++j) {
            v8f z = {0.f,0.f,0.f,0.f,0.f,0.f,0.f,0.f};
            acc[i][j] = z;
        }

    for (int kk = 0; kk < IM; kk += KB_) {
        const uv4* src = (const uv4*)(H + (size_t)(rowStart + arow) * IM + kk + ahalf * 16);
        *(uv4*)&aT[arow][ahalf * 16]     = src[0];
        *(uv4*)&aT[arow][ahalf * 16 + 8] = src[1];
        stageB(bT, wdE + (size_t)(kk + bkp) * DM + colStart + bn0, DM, bkp, bn0);
        __syncthreads();
        v16bf af[2], bf[2];
        #pragma unroll
        for (int i = 0; i < 2; ++i) {
            const __bf16* ap = &aT[wr * 32 + i * 16 + lm][lsel * 16];
            af[i] = frag16(ap, ap + 8);
        }
        #pragma unroll
        for (int j = 0; j < 2; ++j) {
            const __bf16* bp = &bT[wc * 32 + j * 16 + lm][lsel * 16];
            bf[j] = frag16(bp, bp + 8);
        }
        #pragma unroll
        for (int i = 0; i < 2; ++i)
            #pragma unroll
            for (int j = 0; j < 2; ++j)
                acc[i][j] = __builtin_amdgcn_wmma_f32_16x16x32_bf16(
                    false, af[i], false, bf[j], (short)0, acc[i][j], false, false);
        __syncthreads();
    }
    #pragma unroll
    for (int i = 0; i < 2; ++i)
        #pragma unroll
        for (int j = 0; j < 2; ++j)
            #pragma unroll
            for (int r = 0; r < 8; ++r) {
                int m  = r + 8 * lsel;
                int gr = rowStart + wr * 32 + i * 16 + m;
                int gc = colStart + wc * 32 + j * 16 + lm;
                outp[(size_t)gr * DM + gc] = acc[i][j][r];
            }
}

// ---------------- combine: out[t] = w0*op[slot0] + w1*op[slot1] ----------------
__global__ __launch_bounds__(256) void k_combine(const float* __restrict__ outp,
                                                 const int* __restrict__ slotOf,
                                                 const float2* __restrict__ tkW,
                                                 float* __restrict__ out) {
    size_t idx = (size_t)blockIdx.x * 256 + threadIdx.x;
    int t = (int)(idx >> 10);
    int d = (int)(idx & 1023);
    float2 w = tkW[t];
    int s0 = slotOf[2 * t], s1 = slotOf[2 * t + 1];
    out[idx] = w.x * outp[(size_t)s0 * DM + d] + w.y * outp[(size_t)s1 * DM + d];
}

extern "C" void kernel_launch(void* const* d_in, const int* in_sizes, int n_in,
                              void* d_out, int out_size, void* d_ws, size_t ws_size,
                              hipStream_t stream) {
    (void)in_sizes; (void)n_in; (void)out_size; (void)ws_size;
    const float* x     = (const float*)d_in[0];
    const float* wgate = (const float*)d_in[1];
    const float* wg    = (const float*)d_in[2];
    const float* wu    = (const float*)d_in[3];
    const float* wd    = (const float*)d_in[4];
    float* out = (float*)d_out;

    // workspace carve-out (~231 MB total)
    char* ws = (char*)d_ws;
    size_t off = 0;
    auto take = [&](size_t bytes) -> char* {
        char* p = ws + off;
        off = (off + bytes + 255) & ~(size_t)255;
        return p;
    };
    int*    cnt    = (int*)take(NE * 4);
    int*    cursor = (int*)take(NE * 4);
    int*    padOff = (int*)take((NE + 1) * 4);
    float*  probs  = (float*)take((size_t)NT * NE * 4);
    int2*   tkIdx  = (int2*)take((size_t)NT * 8);
    float2* tkW    = (float2*)take((size_t)NT * 8);
    int*    rowTok = (int*)take((size_t)PP * 4);
    int*    slotOf = (int*)take((size_t)NT * 2 * 4);
    __bf16* xb     = (__bf16*)take((size_t)NT * DM * 2);
    __bf16* Hbuf   = (__bf16*)take((size_t)PP * IM * 2);
    float*  outp   = (float*)take((size_t)PP * DM * 4);

    k_init<<<(PP + 255) / 256, 256, 0, stream>>>(cnt, cursor, rowTok);
    k_cvt<<<(NT * DM / 4) / 256, 256, 0, stream>>>(x, xb);
    k_router<<<NT / 8, 256, 0, stream>>>(x, wgate, probs, tkIdx, tkW, cnt);
    k_scan<<<1, 1, 0, stream>>>(cnt, padOff);
    k_aux<<<1, 256, 0, stream>>>(probs, cnt, out + (size_t)NT * DM);
    k_scatter<<<NT / 256, 256, 0, stream>>>(tkIdx, padOff, cursor, rowTok, slotOf);
    dim3 g12(IM / CB_, PP / RB_);
    k_gemm12<<<g12, 256, 0, stream>>>(xb, wg, wu, rowTok, padOff, Hbuf);
    dim3 g3(DM / CB_, PP / RB_);
    k_gemm3<<<g3, 256, 0, stream>>>(Hbuf, wd, padOff, outp);
    k_combine<<<(NT * DM) / 256, 256, 0, stream>>>(outp, slotOf, tkW, out);
}
